// _Block2048_10849087390343
// MI455X (gfx1250) — compile-verified, matrix operand order checked by reading, not run on hardware
//
#include <hip/hip_runtime.h>
#include <hip/hip_bf16.h>

// ---------------------------------------------------------------------------
// Types
// ---------------------------------------------------------------------------
typedef unsigned short u16;
typedef __attribute__((ext_vector_type(4)))  unsigned int u32x4;
typedef __attribute__((ext_vector_type(8)))  int          i32x8;
typedef __attribute__((ext_vector_type(4)))  int          i32x4;
typedef __attribute__((ext_vector_type(16))) __bf16       v16bf;
typedef __attribute__((ext_vector_type(8)))  float        v8f;

union Frag16 { v16bf v; u32x4 q[2]; };

__device__ inline v8f v8zero() {
    v8f z = {0.f,0.f,0.f,0.f,0.f,0.f,0.f,0.f};
    return z;
}

__device__ inline u16 f2bf(float f) {
    unsigned u = __float_as_uint(f);
    u += 0x7fffu + ((u >> 16) & 1u);   // round-to-nearest-even
    return (u16)(u >> 16);
}

__device__ inline v8f wmma_bf16(const Frag16& a, const Frag16& b, v8f c) {
    // D = A(16x32 bf16) * B(32x16 bf16) + C(16x16 f32)
    return __builtin_amdgcn_wmma_f32_16x16x32_bf16(
        false, a.v, false, b.v, (short)0, c, false, false);
}

// One 16x32 bf16 operand fragment from row-major storage, ld in elements.
// lane = {row = lane&15, halfK = lane>>4}; two 8-elem chunks at
// k = halfK*8 and k = 16 + halfK*8 (CDNA5 16-bit A/B VGPR layout).
__device__ inline Frag16 load_frag(const u16* base, int row, int ld, int lane) {
    const int hv = (lane >> 4) & 1;
    const u16* p = base + (size_t)row * ld + hv * 8;
    Frag16 f;
    f.q[0] = *reinterpret_cast<const u32x4*>(p);
    f.q[1] = *reinterpret_cast<const u32x4*>(p + 16);
    return f;
}

// ---------------------------------------------------------------------------
// TDM: 2-D tile (rows x 32 bf16) from row-major global (stride = ld elements)
// into LDS with hardware row padding: pad_interval=16 DWORDs (one 32-elem
// row = 64B), pad_amount=4 DWORDs (8 elems) -> LDS row stride 40 elements,
// matching load_frag's bank-conflict-avoiding layout.
// D# packing per CDNA5 ISA (2 groups, 2-D).
// ---------------------------------------------------------------------------
__device__ inline void tdm_load_2d(unsigned lds_addr, const u16* gptr,
                                   int rows, long long ld_elems) {
    const unsigned long long ga = (unsigned long long)(uintptr_t)gptr;
    u32x4 g0;
    g0[0] = 1u;                                       // count=1, user D#
    g0[1] = lds_addr;                                 // lds_addr [63:32]
    g0[2] = (unsigned)(ga & 0xffffffffull);           // global_addr lo
    g0[3] = (unsigned)((ga >> 32) & 0x01ffffffull)    // global_addr [56:32]
          | (2u << 30);                               // type = 2 (image)
    i32x8 g1;
    g1[0] = (int)((1u << 16)      // data_size = 2 bytes
          |       (1u << 20)      // pad_enable
          |       (3u << 22)      // pad_interval: 16 DWORDs
          |       (3u << 25));    // pad_amount:   4 DWORDs
    g1[1] = (int)(32u << 16);                         // tensor_dim0 lo16 (=32)
    g1[2] = (int)((unsigned)rows << 16);              // dim0 hi=0 | tensor_dim1 lo
    g1[3] = (int)(32u << 16);                         // dim1 hi=0 | tile_dim0 = 32
    g1[4] = (int)(unsigned)rows;                      // tile_dim1 | tile_dim2=0
    g1[5] = (int)(unsigned)(ld_elems & 0xffffffffll); // tensor_dim0_stride lo32
    g1[6] = (int)(unsigned)((ld_elems >> 32) & 0xffffll); // stride hi16 | d1stride=0
    g1[7] = 0;
    i32x4 z4 = {0, 0, 0, 0};
    i32x8 z8 = {0, 0, 0, 0, 0, 0, 0, 0};
    // amdgpu-toolchain (clang-23) 6-arg form: (g0, g1, g2, g3, g_extra, cpol)
    __builtin_amdgcn_tensor_load_to_lds(g0, g1, z4, z4, z8, 0);
}

__device__ inline unsigned lds_off(const void* p) {
    return (unsigned)(uintptr_t)p;    // LDS aperture: addr[31:0] is the offset
}

// ---------------------------------------------------------------------------
// Problem constants
// ---------------------------------------------------------------------------
#define BD     2
#define LSEQ   2048
#define DMODEL 1024
#define NTOK   (BD * LSEQ)      // 4096
#define NHEAD  16
#define FF     2730
#define FFP    2816             // multiple of 128 (N tiles) and 32 (K loop)

// ---------------------------------------------------------------------------
// f32 -> bf16 with zero padding. out [NrowPad x Kpad], in [Nrow x Kcol].
// ---------------------------------------------------------------------------
__global__ __launch_bounds__(256)
void k_convert_pad(const float* __restrict__ in, u16* __restrict__ out,
                   int Nrow, int Kcol, int Kpad, int total) {
    int i = blockIdx.x * 256 + threadIdx.x;
    if (i >= total) return;
    int row = i / Kpad;
    int col = i - row * Kpad;
    float v = (row < Nrow && col < Kcol) ? in[(size_t)row * Kcol + col] : 0.f;
    out[i] = f2bf(v);
}

// ---------------------------------------------------------------------------
// RMSNorm over D=1024, one token per 256-thread block, bf16 output
// ---------------------------------------------------------------------------
__global__ __launch_bounds__(256)
void k_rmsnorm_bf16(const float* __restrict__ x, const float* __restrict__ w,
                    u16* __restrict__ out) {
    const int tok = blockIdx.x;
    const float* xp = x + (size_t)tok * DMODEL;
    float vals[4];
    float ss = 0.f;
#pragma unroll
    for (int i = 0; i < 4; ++i) {
        vals[i] = xp[threadIdx.x + i * 256];
        ss += vals[i] * vals[i];
    }
#pragma unroll
    for (int m = 16; m >= 1; m >>= 1) ss += __shfl_xor(ss, m, 32);
    __shared__ float wsum[8];
    if ((threadIdx.x & 31) == 0) wsum[threadIdx.x >> 5] = ss;
    __syncthreads();
    float tot = 0.f;
#pragma unroll
    for (int i = 0; i < 8; ++i) tot += wsum[i];
    const float rs = rsqrtf(tot * (1.0f / DMODEL) + 1e-6f);
#pragma unroll
    for (int i = 0; i < 4; ++i) {
        int c = threadIdx.x + i * 256;
        out[(size_t)tok * DMODEL + c] = f2bf(vals[i] * rs * w[c]);
    }
}

// ---------------------------------------------------------------------------
// GEMM: C[M,N] = A[M,K] * B[N,K]^T   (bf16 in, f32 WMMA accumulate)
//   EPI 0: outB = bf16(C)                       (wave tile 32x64, block 128x128)
//   EPI 1: outF = C + resid                     (wave tile 32x64, block 128x128)
//   EPI 2: outB = bf16(silu(A*B0^T)*(A*B1^T))   (wave tile 32x32, block 128x64)
// 256 threads = 8 waves (4x2). Double-buffered LDS fed by the Tensor Data
// Mover: wave 0 posts tensor_load_to_lds descriptors for tile k+1 while all
// waves run WMMA on tile k; completion via s_wait_tensorcnt (TENSORcnt).
// ---------------------------------------------------------------------------
template <int EPI>
__global__ __launch_bounds__(256)
void k_gemm_bf16(const u16* __restrict__ A,
                 const u16* __restrict__ B0,
                 const u16* __restrict__ B1,
                 int N, int K,
                 const float* __restrict__ resid,
                 float* __restrict__ outF,
                 u16* __restrict__ outB) {
    constexpr bool DUAL = (EPI == 2);
    constexpr int  CT   = DUAL ? 2 : 4;     // 16-col WMMA tiles per wave
    constexpr int  BN   = 32 * CT;          // block N extent: 64 or 128
    constexpr int  G    = DUAL ? 3 : 2;     // tensor loads per K-step

    __shared__ alignas(16) u16 sA [2][128 * 40];
    __shared__ alignas(16) u16 sB0[2][BN * 40];
    __shared__ alignas(16) u16 sB1[2][DUAL ? BN * 40 : 16];

    const int bn = blockIdx.x * BN;
    const int bm = blockIdx.y * 128;
    const int t = threadIdx.x;
    const int wid = t >> 5;
    const int lane = t & 31;
    const int ln = lane & 15;
    const int wm = (wid & 3) * 32;
    const int wn = (wid >> 2) * (CT * 16);

    const u16* Abase  = A  + (size_t)bm * K;
    const u16* B0base = B0 + (size_t)bn * K;
    const u16* B1base = DUAL ? (B1 + (size_t)bn * K) : B0base;

    v8f acc [2][4];
    v8f acc1[2][4];
#pragma unroll
    for (int i = 0; i < 2; ++i)
#pragma unroll
        for (int j = 0; j < 4; ++j) { acc[i][j] = v8zero(); acc1[i][j] = v8zero(); }

    auto issue = [&](int buf, int k0) {
        if (wid == 0) {
            tdm_load_2d(lds_off(&sA[buf][0]),  Abase  + k0, 128, K);
            tdm_load_2d(lds_off(&sB0[buf][0]), B0base + k0, BN,  K);
            if (DUAL)
                tdm_load_2d(lds_off(&sB1[buf][0]), B1base + k0, BN, K);
        }
    };
    auto compute = [&](const u16* a, const u16* b0, const u16* b1) {
        Frag16 af[2];
#pragma unroll
        for (int i = 0; i < 2; ++i) af[i] = load_frag(a, wm + i * 16 + ln, 40, lane);
#pragma unroll
        for (int j = 0; j < CT; ++j) {
            Frag16 bf = load_frag(b0, wn + j * 16 + ln, 40, lane);
#pragma unroll
            for (int i = 0; i < 2; ++i) acc[i][j] = wmma_bf16(af[i], bf, acc[i][j]);
            if (DUAL) {
                Frag16 bg = load_frag(b1, wn + j * 16 + ln, 40, lane);
#pragma unroll
                for (int i = 0; i < 2; ++i) acc1[i][j] = wmma_bf16(af[i], bg, acc1[i][j]);
            }
        }
    };

    const int nk = K >> 5;                 // K/32 tiles (K % 32 == 0)
    issue(0, 0);                           // prologue: fill buffer 0
    for (int it = 0; it < nk; ++it) {
        const int buf = it & 1;
        __syncthreads();                   // buf^1 free (prev compute done)
        if (it + 1 < nk) {
            issue(buf ^ 1, (it + 1) * 32); // TDM fills next buffer
            if (wid == 0) __builtin_amdgcn_s_wait_tensorcnt(G);  // buf resident
        } else {
            if (wid == 0) __builtin_amdgcn_s_wait_tensorcnt(0);
        }
        __syncthreads();                   // publish buf to all waves
        compute(sA[buf], sB0[buf], sB1[buf]);
    }

    // Epilogue: C layout -> lane holds col (lane&15), rows r + 8*(lane>>4)
    const int hv = (lane >> 4) & 1;
#pragma unroll
    for (int i = 0; i < 2; ++i)
#pragma unroll
        for (int j = 0; j < CT; ++j) {
            const int col  = bn + wn + j * 16 + ln;
            const int row0 = bm + wm + i * 16 + 8 * hv;
#pragma unroll
            for (int r = 0; r < 8; ++r) {
                const size_t idx = (size_t)(row0 + r) * N + col;
                const float v = acc[i][j][r];
                if (EPI == 0) {
                    outB[idx] = f2bf(v);
                } else if (EPI == 1) {
                    outF[idx] = v + resid[idx];
                } else {
                    const float s = v / (1.f + __expf(-v));   // silu
                    outB[idx] = f2bf(s * acc1[i][j][r]);
                }
            }
        }
}

// ---------------------------------------------------------------------------
// Flash attention, 64-key tiles (bf16 WMMA, f32 online softmax).
// qkv: [NTOK, 3072] bf16; q|k|v at col offsets 0|1024|2048, head h at +h*64.
// out: bf16 [NTOK, 1024]. grid = (L/64, B*H); 128 threads = 4 waves,
// each wave owns 16 q-rows; 16 WMMA per 64-key iteration.
// ---------------------------------------------------------------------------
__global__ __launch_bounds__(128)
void k_flash_attn(const u16* __restrict__ qkv, u16* __restrict__ out) {
    const int bh = blockIdx.y;
    const int b  = bh >> 4;
    const int h  = bh & 15;
    const int q0 = blockIdx.x * 64;
    const int t = threadIdx.x;
    const int wid = t >> 5;
    const int lane = t & 31;
    const int hv = (lane >> 4) & 1;
    const int ln = lane & 15;
    const int LD = 3 * DMODEL;
    const size_t tokbase = (size_t)b * LSEQ;

    const u16* Qb = qkv + tokbase * LD + h * 64;
    const u16* Kb = qkv + tokbase * LD + DMODEL + h * 64;
    const u16* Vb = qkv + tokbase * LD + 2 * DMODEL + h * 64;

    __shared__ alignas(16) u16 sK [64 * 72];      // K tile  [64 keys][64 dk]
    __shared__ alignas(16) u16 sVt[64 * 72];      // V^T     [64 dk][64 keys]
    __shared__ alignas(16) u16 sP [4][16 * 72];   // per-wave P [16 q][64 keys]

    // Q fragments (2 k-steps over DK=64), fixed for whole kernel
    Frag16 qf[2];
    {
        const int qrow = q0 + wid * 16 + ln;
        const u16* pr = Qb + (size_t)qrow * LD;
#pragma unroll
        for (int ks = 0; ks < 2; ++ks) {
            const u16* p = pr + ks * 32 + hv * 8;
            qf[ks].q[0] = *(const u32x4*)p;
            qf[ks].q[1] = *(const u32x4*)(p + 16);
        }
    }

    v8f o[4];
#pragma unroll
    for (int d = 0; d < 4; ++d) o[d] = v8zero();
    float mrow[8], lrow[8];
#pragma unroll
    for (int r = 0; r < 8; ++r) { mrow[r] = -1e30f; lrow[r] = 0.f; }

    for (int kt = 0; kt < LSEQ / 64; ++kt) {
        // cooperative stage: K row-major, V transposed. 128 threads x 32 elems.
        {
            const int r = t >> 1, cb = (t & 1) * 32;
            const u16* gk = Kb + (size_t)(kt * 64 + r) * LD + cb;
            u32x4 kv[4];
#pragma unroll
            for (int c = 0; c < 4; ++c) kv[c] = *(const u32x4*)(gk + c * 8);
#pragma unroll
            for (int c = 0; c < 4; ++c) *(u32x4*)(sK + r * 72 + cb + c * 8) = kv[c];

            const u16* gv = Vb + (size_t)(kt * 64 + r) * LD + cb;
            u16 tmp[32];
#pragma unroll
            for (int c = 0; c < 4; ++c) *(u32x4*)(tmp + c * 8) = *(const u32x4*)(gv + c * 8);
#pragma unroll
            for (int e = 0; e < 32; ++e) sVt[(cb + e) * 72 + r] = tmp[e];

            if (kt + 1 < LSEQ / 64) {     // L2 prefetch of next tile
                __builtin_prefetch(gk + (size_t)64 * LD, 0, 3);
                __builtin_prefetch(gv + (size_t)64 * LD, 0, 3);
            }
        }
        __syncthreads();

        // S = Q K^T : four 16x16 accumulators over 64 keys
        v8f s[4];
#pragma unroll
        for (int j = 0; j < 4; ++j) s[j] = v8zero();
#pragma unroll
        for (int ks = 0; ks < 2; ++ks)
#pragma unroll
            for (int j = 0; j < 4; ++j) {
                Frag16 kf = load_frag(sK + ks * 32, j * 16 + ln, 72, lane);
                s[j] = wmma_bf16(qf[ks], kf, s[j]);
            }

        // online softmax over this 64-key tile (rows = r + 8*hv per lane)
        float p[4][8], mnew[8], alpha[8];
#pragma unroll
        for (int r = 0; r < 8; ++r) {
            float mx = -1e30f;
#pragma unroll
            for (int j = 0; j < 4; ++j) { p[j][r] = s[j][r] * 0.125f; mx = fmaxf(mx, p[j][r]); }
            mx = fmaxf(mx, __shfl_xor(mx, 1, 32));
            mx = fmaxf(mx, __shfl_xor(mx, 2, 32));
            mx = fmaxf(mx, __shfl_xor(mx, 4, 32));
            mx = fmaxf(mx, __shfl_xor(mx, 8, 32));
            mnew[r]  = fmaxf(mrow[r], mx);
            alpha[r] = __expf(mrow[r] - mnew[r]);
            mrow[r]  = mnew[r];
        }
#pragma unroll
        for (int r = 0; r < 8; ++r) {
            float sm = 0.f;
#pragma unroll
            for (int j = 0; j < 4; ++j) {
                const float e = __expf(p[j][r] - mnew[r]);
                sm += e;
                sP[wid][(8 * hv + r) * 72 + j * 16 + ln] = f2bf(e);
            }
            sm += __shfl_xor(sm, 1, 32);
            sm += __shfl_xor(sm, 2, 32);
            sm += __shfl_xor(sm, 4, 32);
            sm += __shfl_xor(sm, 8, 32);
            lrow[r] = lrow[r] * alpha[r] + sm;
        }
#pragma unroll
        for (int d = 0; d < 4; ++d)
#pragma unroll
            for (int r = 0; r < 8; ++r) o[d][r] *= alpha[r];

        // O += P * V  (two A-frags of P; V^T rows are dk columns)
        Frag16 pf[2];
        pf[0] = load_frag(sP[wid],      ln, 72, lane);
        pf[1] = load_frag(sP[wid] + 32, ln, 72, lane);
#pragma unroll
        for (int d = 0; d < 4; ++d)
#pragma unroll
            for (int ks = 0; ks < 2; ++ks) {
                Frag16 vf = load_frag(sVt + ks * 32, d * 16 + ln, 72, lane);
                o[d] = wmma_bf16(pf[ks], vf, o[d]);
            }
        __syncthreads();
    }

    // finalize: divide by denominator, store bf16 head-concatenated
#pragma unroll
    for (int d = 0; d < 4; ++d)
#pragma unroll
        for (int r = 0; r < 8; ++r) {
            const int row = q0 + wid * 16 + 8 * hv + r;
            const float v = o[d][r] / lrow[r];
            out[(tokbase + row) * DMODEL + h * 64 + d * 16 + ln] = f2bf(v);
        }
}

// ---------------------------------------------------------------------------
// Host launcher
// ---------------------------------------------------------------------------
extern "C" void kernel_launch(void* const* d_in, const int* in_sizes, int n_in,
                              void* d_out, int out_size, void* d_ws, size_t ws_size,
                              hipStream_t stream) {
    const float* x     = (const float*)d_in[0];  // [2,2048,1024]
    const float* w_qkv = (const float*)d_in[1];  // [3072,1024]
    const float* w_o   = (const float*)d_in[2];  // [1024,1024]
    const float* n1_w  = (const float*)d_in[3];
    const float* n2_w  = (const float*)d_in[4];
    const float* w1    = (const float*)d_in[5];  // [2730,1024]
    const float* w2    = (const float*)d_in[6];  // [1024,2730]
    const float* w3    = (const float*)d_in[7];  // [2730,1024]
    float* outp = (float*)d_out;                 // [2,2048,1024]

    size_t off = 0;
    auto alloc = [&](size_t bytes) -> void* {
        off = (off + 255) & ~(size_t)255;
        void* p = (char*)d_ws + off;
        off += bytes;
        return p;
    };
    u16*   xn1    = (u16*)  alloc((size_t)NTOK * DMODEL * 2);
    u16*   wqkv_b = (u16*)  alloc((size_t)3 * DMODEL * DMODEL * 2);
    u16*   wo_b   = (u16*)  alloc((size_t)DMODEL * DMODEL * 2);
    u16*   w1_b   = (u16*)  alloc((size_t)FFP * DMODEL * 2);
    u16*   w3_b   = (u16*)  alloc((size_t)FFP * DMODEL * 2);
    u16*   w2_b   = (u16*)  alloc((size_t)DMODEL * FFP * 2);
    u16*   qkv_b  = (u16*)  alloc((size_t)NTOK * 3 * DMODEL * 2);
    u16*   attn_b = (u16*)  alloc((size_t)NTOK * DMODEL * 2);
    float* x1     = (float*)alloc((size_t)NTOK * DMODEL * 4);
    u16*   xn2    = (u16*)  alloc((size_t)NTOK * DMODEL * 2);
    u16*   g_b    = (u16*)  alloc((size_t)NTOK * FFP * 2);
    (void)ws_size; (void)in_sizes; (void)n_in; (void)out_size;

    // 1) weight conversion f32 -> bf16 (+ zero padding for FF dims)
    {
        int tot;
        tot = 3 * DMODEL * DMODEL;
        k_convert_pad<<<tot / 256, 256, 0, stream>>>(w_qkv, wqkv_b, 3 * DMODEL, DMODEL, DMODEL, tot);
        tot = DMODEL * DMODEL;
        k_convert_pad<<<tot / 256, 256, 0, stream>>>(w_o, wo_b, DMODEL, DMODEL, DMODEL, tot);
        tot = FFP * DMODEL;
        k_convert_pad<<<tot / 256, 256, 0, stream>>>(w1, w1_b, FF, DMODEL, DMODEL, tot);
        k_convert_pad<<<tot / 256, 256, 0, stream>>>(w3, w3_b, FF, DMODEL, DMODEL, tot);
        tot = DMODEL * FFP;
        k_convert_pad<<<tot / 256, 256, 0, stream>>>(w2, w2_b, DMODEL, FF, FFP, tot);
    }

    // 2) xn1 = bf16(rmsnorm(x, n1_w))
    k_rmsnorm_bf16<<<NTOK, 256, 0, stream>>>(x, n1_w, xn1);

    // 3) qkv = xn1 @ w_qkv^T         [4096 x 3072]
    k_gemm_bf16<0><<<dim3(3 * DMODEL / 128, NTOK / 128), 256, 0, stream>>>(
        xn1, wqkv_b, nullptr, 3 * DMODEL, DMODEL, nullptr, nullptr, qkv_b);

    // 4) attention (flash, 64-key tiles)
    k_flash_attn<<<dim3(LSEQ / 64, BD * NHEAD), 128, 0, stream>>>(qkv_b, attn_b);

    // 5) x1 = attn @ w_o^T + x       [4096 x 1024] f32
    k_gemm_bf16<1><<<dim3(DMODEL / 128, NTOK / 128), 256, 0, stream>>>(
        attn_b, wo_b, nullptr, DMODEL, DMODEL, x, x1, nullptr);

    // 6) xn2 = bf16(rmsnorm(x1, n2_w))
    k_rmsnorm_bf16<<<NTOK, 256, 0, stream>>>(x1, n2_w, xn2);

    // 7) g = silu(xn2 @ w1^T) * (xn2 @ w3^T)   [4096 x 2816] bf16 (fused dual-B)
    k_gemm_bf16<2><<<dim3(FFP / 64, NTOK / 128), 256, 0, stream>>>(
        xn2, w1_b, w3_b, FFP, DMODEL, nullptr, nullptr, g_b);

    // 8) out = g @ w2^T + x1         [4096 x 1024] f32
    k_gemm_bf16<1><<<dim3(DMODEL / 128, NTOK / 128), 256, 0, stream>>>(
        g_b, w2_b, nullptr, DMODEL, FFP, x1, outp, nullptr);
}
